// GCN_Model_reg_78993038508802
// MI455X (gfx1250) — matrix operand order repaired and checked by reading, not compiled, not run on hardware
//
#include <hip/hip_runtime.h>
#include <hip/hip_bf16.h>

// Problem sizes (fixed by the reference)
#define NN   16384
#define EE   65536
#define DD   64
#define GG   512
#define F1S  256
#define F2S  128
#define KTOT 130          // 4096/32 k-tiles + 2 bias tiles (K = 4160)
#define NCHK 65           // k-chunks of 2 tiles each
#define EPS  1e-5f

typedef __attribute__((ext_vector_type(16))) _Float16 v16h;
typedef __attribute__((ext_vector_type(8)))  _Float16 v8h;
typedef __attribute__((ext_vector_type(8)))  float    v8f;

// ---------------------------------------------------------------- utilities
__global__ void fill_zero(float* __restrict__ p, int n) {
    int i = blockIdx.x * blockDim.x + threadIdx.x;
    if (i < n) p[i] = 0.0f;
}

// ----------------------------------------------------- lin0: h = relu(xW^T+b)
__global__ __launch_bounds__(256) void lin0_relu(
    const float* __restrict__ x, const float* __restrict__ w,
    const float* __restrict__ b, float* __restrict__ h) {
    int tid = blockIdx.x * blockDim.x + threadIdx.x;   // N*64 threads
    int c = tid & 63, n = tid >> 6;
    float acc = b[c];
    #pragma unroll
    for (int k = 0; k < 16; ++k) acc += x[n * 16 + k] * w[c * 16 + k];
    h[(size_t)n * DD + c] = fmaxf(acc, 0.0f);
}

// ---------------------- per-edge: feat = relu(edge_attr@W1^T+b1), gather x_src
__global__ __launch_bounds__(256) void edge_prep(
    const float* __restrict__ ea, const float* __restrict__ w1,
    const float* __restrict__ b1, const float* __restrict__ h,
    const int* __restrict__ src,
    _Float16* __restrict__ feat_h, _Float16* __restrict__ xs_h) {
    int tid = blockIdx.x * blockDim.x + threadIdx.x;   // E*64 threads
    int c = tid & 63, e = tid >> 6;
    float acc = b1[c];
    #pragma unroll
    for (int k = 0; k < 6; ++k) acc += ea[e * 6 + k] * w1[c * 6 + k];
    feat_h[(size_t)e * DD + c] = (_Float16)fmaxf(acc, 0.0f);
    xs_h  [(size_t)e * DD + c] = (_Float16)h[(size_t)src[e] * DD + c];
}

// ------------- pack B into WMMA fragment-major order: [kt][nt][lane][j] (f16)
// B[k,o] = (k<4096) ? W2[(k&63)*64+o , k>>6] : b2[(k-4096)*64+o]
__global__ __launch_bounds__(256) void pack_B(
    const float* __restrict__ W2, const float* __restrict__ b2,
    _Float16* __restrict__ Bp) {
    int idx = blockIdx.x * blockDim.x + threadIdx.x;   // 130*4*32*16 = 266240
    if (idx >= KTOT * 4 * 32 * 16) return;
    int j    = idx & 15;
    int lane = (idx >> 4) & 31;
    int nt   = (idx >> 9) & 3;
    int kt   = idx >> 11;
    int k = kt * 32 + ((lane < 16) ? 0 : 8) + ((j < 8) ? j : 8 + j);      // ISA B 32x16 lane map (table order)
    int o = nt * 16 + (lane & 15);
    float v = (k < 4096) ? W2[(size_t)((k & 63) * 64 + o) * 64 + (k >> 6)]
                         : b2[(k - 4096) * 64 + o];
    Bp[idx] = (_Float16)v;
}

// --------------------------------- the big one: [E,4160]x[4160,64] f16 WMMA
// A generated on the fly: A[e,k] = feat[e,k>>6] * xs[e,k&63] (bias tiles: f=1).
// B is staged per-block through LDS with double-buffered async global->LDS
// copies (ASYNCcnt), so all 8 waves share one fetch of each fragment block.
__global__ __launch_bounds__(256) void nnconv_wmma(
    const _Float16* __restrict__ xs_h, const _Float16* __restrict__ feat_h,
    const _Float16* __restrict__ Bp, const int* __restrict__ dst,
    float* __restrict__ aggr) {
    __shared__ __align__(32) _Float16 sB[2][4096];     // [buf][kt_sub*2048 + frag]

    const int tid  = threadIdx.x;
    const int lane = tid & 31;
    const int wave = (blockIdx.x * blockDim.x + tid) >> 5;
    const int eb   = wave * 16;                 // 16 edges per wave
    const int e    = eb + (lane & 15);          // A row m = lane%16
    const _Float16* xr = xs_h   + (size_t)e * DD;
    const _Float16* fr = feat_h + (size_t)e * DD;
    const int kab = (lane < 16) ? 0 : 8;        // ISA A 16x32 lane k-base

    // Hoist the only 4 x-chunks this lane ever needs (64 B) into registers.
    v8h xA0 = *(const v8h*)(xr + kab);          // even kt: K = kab+0..7
    v8h xB0 = *(const v8h*)(xr + kab + 16);     // even kt: K = kab+16..23
    v8h xA1 = *(const v8h*)(xr + 32 + kab);     // odd  kt
    v8h xB1 = *(const v8h*)(xr + 48 + kab);

    v8f acc0 = {}, acc1 = {}, acc2 = {}, acc3 = {};

    // LDS byte addresses for this thread's async-copy destination (32 B each).
    const unsigned lds0 = (unsigned)(uintptr_t)(&sB[0][0]) + (unsigned)tid * 32u;
    const unsigned lds1 = (unsigned)(uintptr_t)(&sB[1][0]) + (unsigned)tid * 32u;

    // Stage chunk 0 into buffer 0 (2 x b128 per thread = 8 KB per block).
    {
        unsigned long long g =
            (unsigned long long)(uintptr_t)(Bp + (size_t)tid * 16);
        asm volatile("global_load_async_to_lds_b128 %0, %1, off\n\t"
                     "global_load_async_to_lds_b128 %0, %1, off offset:16"
                     :: "v"(lds0), "v"(g) : "memory");
    }

    for (int c = 0; c < NCHK; ++c) {
        const int buf = c & 1;
        if (c + 1 < NCHK) {                     // prefetch next chunk
            unsigned long long g = (unsigned long long)(uintptr_t)
                (Bp + (size_t)(c + 1) * 4096 + (size_t)tid * 16);
            asm volatile("global_load_async_to_lds_b128 %0, %1, off\n\t"
                         "global_load_async_to_lds_b128 %0, %1, off offset:16"
                         :: "v"(buf ? lds0 : lds1), "v"(g) : "memory");
            // in-order completion: cnt<=2 => chunk c fully in LDS
            asm volatile("s_wait_asynccnt 0x2" ::: "memory");
        } else {
            asm volatile("s_wait_asynccnt 0x0" ::: "memory");
        }
        __syncthreads();                        // whole block sees chunk c

        // A fragments for the two k-tiles of this chunk (feat scalar is
        // constant per tile; bias tiles use feat = 1).
        _Float16 f = (c < 64) ? fr[c] : (_Float16)1.0f;
        v16h A0, A1;
        #pragma unroll
        for (int t = 0; t < 8; ++t) {
            A0[t] = f * xA0[t]; A0[t + 8] = f * xB0[t];
            A1[t] = f * xA1[t]; A1[t + 8] = f * xB1[t];
        }

        const _Float16* b0 = &sB[buf][0]    + lane * 16;   // kt_sub = 0
        const _Float16* b1 = &sB[buf][2048] + lane * 16;   // kt_sub = 1
        v16h B;
        B = *(const v16h*)(b0);
        acc0 = __builtin_amdgcn_wmma_f32_16x16x32_f16(false, A0, false, B, (short)0, acc0, false, false);
        B = *(const v16h*)(b0 + 512);
        acc1 = __builtin_amdgcn_wmma_f32_16x16x32_f16(false, A0, false, B, (short)0, acc1, false, false);
        B = *(const v16h*)(b0 + 1024);
        acc2 = __builtin_amdgcn_wmma_f32_16x16x32_f16(false, A0, false, B, (short)0, acc2, false, false);
        B = *(const v16h*)(b0 + 1536);
        acc3 = __builtin_amdgcn_wmma_f32_16x16x32_f16(false, A0, false, B, (short)0, acc3, false, false);
        B = *(const v16h*)(b1);
        acc0 = __builtin_amdgcn_wmma_f32_16x16x32_f16(false, A1, false, B, (short)0, acc0, false, false);
        B = *(const v16h*)(b1 + 512);
        acc1 = __builtin_amdgcn_wmma_f32_16x16x32_f16(false, A1, false, B, (short)0, acc1, false, false);
        B = *(const v16h*)(b1 + 1024);
        acc2 = __builtin_amdgcn_wmma_f32_16x16x32_f16(false, A1, false, B, (short)0, acc2, false, false);
        B = *(const v16h*)(b1 + 1536);
        acc3 = __builtin_amdgcn_wmma_f32_16x16x32_f16(false, A1, false, B, (short)0, acc3, false, false);

        __syncthreads();   // protect buf being overwritten next iteration
    }

    // C/D layout: VGPR r, lane L -> M = r + (L<16?0:8), N = L%16
    const int mbase = (lane < 16) ? 0 : 8;
    const int col   = lane & 15;
    #pragma unroll
    for (int r = 0; r < 8; ++r) {
        int d = dst[eb + mbase + r];
        float* ap = aggr + (size_t)d * DD + col;
        atomicAdd(ap +  0, acc0[r]);
        atomicAdd(ap + 16, acc1[r]);
        atomicAdd(ap + 32, acc2[r]);
        atomicAdd(ap + 48, acc3[r]);
    }
}

// ------------------------ node update: root+bias, relu, BN, single-step GRU
__global__ __launch_bounds__(256) void node_update(
    const float* __restrict__ aggr, const float* __restrict__ h_in,
    float* __restrict__ h_out, const float* __restrict__ root_w,
    const float* __restrict__ conv_b, const float* __restrict__ bn,
    const float* __restrict__ wih, const float* __restrict__ whh,
    const float* __restrict__ bih, const float* __restrict__ bhh) {
    __shared__ float sh[4][DD];
    __shared__ float sm[4][DD];
    int c  = threadIdx.x & 63;
    int li = threadIdx.x >> 6;
    int n  = blockIdx.x * 4 + li;
    float hv = h_in[(size_t)n * DD + c];
    sh[li][c] = hv;
    __syncthreads();
    float acc = aggr[(size_t)n * DD + c] + conv_b[c];
    #pragma unroll 8
    for (int i = 0; i < DD; ++i) acc += sh[li][i] * root_w[i * DD + c]; // h@root
    acc = fmaxf(acc, 0.0f);
    acc = (acc - bn[128 + c]) * rsqrtf(bn[192 + c] + EPS) * bn[c] + bn[64 + c];
    sm[li][c] = acc;
    __syncthreads();
    float ir = bih[c], iz = bih[64 + c], in_ = bih[128 + c];
    float hr = bhh[c], hz = bhh[64 + c], hn  = bhh[128 + c];
    #pragma unroll 4
    for (int i = 0; i < DD; ++i) {
        float mi = sm[li][i], hi = sh[li][i];
        ir  += mi * wih[c * DD + i];
        iz  += mi * wih[(64 + c) * DD + i];
        in_ += mi * wih[(128 + c) * DD + i];
        hr  += hi * whh[c * DD + i];
        hz  += hi * whh[(64 + c) * DD + i];
        hn  += hi * whh[(128 + c) * DD + i];
    }
    float r = 1.0f / (1.0f + __expf(-(ir + hr)));
    float z = 1.0f / (1.0f + __expf(-(iz + hz)));
    float nn = tanhf(in_ + r * hn);
    h_out[(size_t)n * DD + c] = (1.0f - z) * nn + z * hv;
}

// ---------------------------------------- GraphConv scatter: aggr[dst]+=h[src]
__global__ __launch_bounds__(256) void gc_scatter(
    const float* __restrict__ h, const int* __restrict__ src,
    const int* __restrict__ dst, float* __restrict__ aggr) {
    int tid = blockIdx.x * blockDim.x + threadIdx.x;   // E*64 threads
    int c = tid & 63, e = tid >> 6;
    atomicAdd(&aggr[(size_t)dst[e] * DD + c], h[(size_t)src[e] * DD + c]);
}

// ------- GraphConv node part + BN + global_add_pool into pooled[G,64]
__global__ __launch_bounds__(256) void gc_node_pool(
    const float* __restrict__ aggr, const float* __restrict__ h,
    const float* __restrict__ rel_w, const float* __restrict__ rel_b,
    const float* __restrict__ root_w, const float* __restrict__ bnl,
    const int* __restrict__ batch, float* __restrict__ pooled) {
    __shared__ float sh[4][DD];
    __shared__ float sa[4][DD];
    int c  = threadIdx.x & 63;
    int li = threadIdx.x >> 6;
    int n  = blockIdx.x * 4 + li;
    sh[li][c] = h[(size_t)n * DD + c];
    sa[li][c] = aggr[(size_t)n * DD + c];
    __syncthreads();
    float acc = rel_b[c];
    #pragma unroll 8
    for (int i = 0; i < DD; ++i)
        acc += sa[li][i] * rel_w[c * DD + i] + sh[li][i] * root_w[c * DD + i];
    acc = fmaxf(acc, 0.0f);
    acc = (acc - bnl[128 + c]) * rsqrtf(bnl[192 + c] + EPS) * bnl[c] + bnl[64 + c];
    atomicAdd(&pooled[(size_t)batch[n] * DD + c], acc);
}

// ----------------------------------------------------------- FC head kernels
__global__ __launch_bounds__(256) void fc1_k(
    const float* __restrict__ pooled, const float* __restrict__ w,
    const float* __restrict__ b, const float* __restrict__ bnp,
    float* __restrict__ f1) {
    __shared__ float sp[DD];
    int g = blockIdx.x, o = threadIdx.x;               // 256 threads
    if (o < DD) sp[o] = pooled[(size_t)g * DD + o];
    __syncthreads();
    float acc = b[o];
    #pragma unroll 8
    for (int i = 0; i < DD; ++i) acc += sp[i] * w[o * DD + i];
    acc = fmaxf(acc, 0.0f);
    acc = (acc - bnp[512 + o]) * rsqrtf(bnp[768 + o] + EPS) * bnp[o] + bnp[256 + o];
    f1[(size_t)g * F1S + o] = acc;
}

__global__ __launch_bounds__(128) void fc2_k(
    const float* __restrict__ f1, const float* __restrict__ w,
    const float* __restrict__ b, const float* __restrict__ bnp,
    float* __restrict__ f2) {
    __shared__ float sp[F1S];
    int g = blockIdx.x, o = threadIdx.x;               // 128 threads
    sp[o] = f1[(size_t)g * F1S + o];
    sp[o + 128] = f1[(size_t)g * F1S + o + 128];
    __syncthreads();
    float acc = b[o];
    #pragma unroll 8
    for (int i = 0; i < F1S; ++i) acc += sp[i] * w[o * F1S + i];
    acc = fmaxf(acc, 0.0f);
    acc = (acc - bnp[256 + o]) * rsqrtf(bnp[384 + o] + EPS) * bnp[o] + bnp[128 + o];
    f2[(size_t)g * F2S + o] = acc;
}

__global__ __launch_bounds__(256) void fcl_k(
    const float* __restrict__ f2, const float* __restrict__ w,
    const float* __restrict__ b, float* __restrict__ out) {
    int g = blockIdx.x * blockDim.x + threadIdx.x;
    if (g >= GG) return;
    float acc = b[0];
    #pragma unroll 8
    for (int i = 0; i < F2S; ++i) acc += f2[(size_t)g * F2S + i] * w[i];
    out[g] = acc;
}

// ---------------------------------------------------------------------------
extern "C" void kernel_launch(void* const* d_in, const int* in_sizes, int n_in,
                              void* d_out, int out_size, void* d_ws, size_t ws_size,
                              hipStream_t stream) {
    const float* x        = (const float*)d_in[0];
    const float* ea       = (const float*)d_in[1];
    const float* lin0_w   = (const float*)d_in[2];
    const float* lin0_b   = (const float*)d_in[3];
    const float* mlp1_w   = (const float*)d_in[4];
    const float* mlp1_b   = (const float*)d_in[5];
    const float* mlp2_w   = (const float*)d_in[6];
    const float* mlp2_b   = (const float*)d_in[7];
    const float* root_w   = (const float*)d_in[8];
    const float* conv_b   = (const float*)d_in[9];
    const float* bn       = (const float*)d_in[10];
    const float* gru_wih  = (const float*)d_in[11];
    const float* gru_whh  = (const float*)d_in[12];
    const float* gru_bih  = (const float*)d_in[13];
    const float* gru_bhh  = (const float*)d_in[14];
    const float* gc_rel_w = (const float*)d_in[15];
    const float* gc_rel_b = (const float*)d_in[16];
    const float* gc_root_w= (const float*)d_in[17];
    const float* bnl      = (const float*)d_in[18];
    const float* fc1_w    = (const float*)d_in[19];
    const float* fc1_b    = (const float*)d_in[20];
    const float* fcbn1    = (const float*)d_in[21];
    const float* fc2_w    = (const float*)d_in[22];
    const float* fc2_b    = (const float*)d_in[23];
    const float* fcbn2    = (const float*)d_in[24];
    const float* fcl_w    = (const float*)d_in[25];
    const float* fcl_b    = (const float*)d_in[26];
    const int*   eidx     = (const int*)d_in[27];
    const int*   batch    = (const int*)d_in[28];
    const int* src = eidx;
    const int* dst = eidx + EE;

    // workspace carve-up (256B aligned)
    char* ws = (char*)d_ws;
    auto carve = [&](size_t bytes) {
        char* p = ws; ws += (bytes + 255) & ~(size_t)255; return p;
    };
    float*    h0     = (float*)   carve((size_t)NN * DD * 4);
    float*    h1     = (float*)   carve((size_t)NN * DD * 4);
    float*    aggr   = (float*)   carve((size_t)NN * DD * 4);
    _Float16* feat_h = (_Float16*)carve((size_t)EE * DD * 2);
    _Float16* xs_h   = (_Float16*)carve((size_t)EE * DD * 2);
    _Float16* Bp     = (_Float16*)carve((size_t)KTOT * 4 * 32 * 16 * 2);
    float*    pooled = (float*)   carve((size_t)GG * DD * 4);
    float*    f1     = (float*)   carve((size_t)GG * F1S * 4);
    float*    f2     = (float*)   carve((size_t)GG * F2S * 4);

    lin0_relu<<<(NN * DD) / 256, 256, 0, stream>>>(x, lin0_w, lin0_b, h0);

    float* hin = h0; float* hout = h1;
    for (int l = 0; l < 2; ++l) {
        edge_prep<<<(EE * DD) / 256, 256, 0, stream>>>(
            ea, mlp1_w + (size_t)l * DD * 6, mlp1_b + (size_t)l * DD,
            hin, src, feat_h, xs_h);
        pack_B<<<(KTOT * 4 * 32 * 16) / 256, 256, 0, stream>>>(
            mlp2_w + (size_t)l * 4096 * DD, mlp2_b + (size_t)l * 4096, Bp);
        fill_zero<<<(NN * DD) / 256, 256, 0, stream>>>(aggr, NN * DD);
        nnconv_wmma<<<(EE / 16) / 8, 256, 0, stream>>>(xs_h, feat_h, Bp, dst, aggr);
        node_update<<<NN / 4, 256, 0, stream>>>(
            aggr, hin, hout,
            root_w + (size_t)l * DD * DD, conv_b + (size_t)l * DD,
            bn + (size_t)l * 4 * DD,
            gru_wih + (size_t)l * 3 * DD * DD, gru_whh + (size_t)l * 3 * DD * DD,
            gru_bih + (size_t)l * 3 * DD, gru_bhh + (size_t)l * 3 * DD);
        float* t = hin; hin = hout; hout = t;
    }

    // GraphConv + BN + pooling
    fill_zero<<<(NN * DD) / 256, 256, 0, stream>>>(aggr, NN * DD);
    gc_scatter<<<(EE * DD) / 256, 256, 0, stream>>>(hin, src, dst, aggr);
    fill_zero<<<(GG * DD) / 256, 256, 0, stream>>>(pooled, GG * DD);
    gc_node_pool<<<NN / 4, 256, 0, stream>>>(aggr, hin, gc_rel_w, gc_rel_b,
                                             gc_root_w, bnl, batch, pooled);

    // FC head
    fc1_k<<<GG, 256, 0, stream>>>(pooled, fc1_w, fc1_b, fcbn1, f1);
    fc2_k<<<GG, 128, 0, stream>>>(f1, fc2_w, fc2_b, fcbn2, f2);
    fcl_k<<<(GG + 255) / 256, 256, 0, stream>>>(f2, fcl_w, fcl_b, (float*)d_out);
}
